// NeuralTensorLayer_3358664426252
// MI455X (gfx1250) — matrix operand order repaired and checked by reading, not compiled, not run on hardware
//
#include <hip/hip_runtime.h>
#include <hip/hip_bf16.h>
#include <math.h>

// Problem constants (from reference): B=4096, D=1024, K=32
#define NB 4096
#define ND 1024
#define NK 32

typedef __bf16 bf16;
typedef __attribute__((ext_vector_type(16))) bf16 v16bf;
typedef __attribute__((ext_vector_type(8)))  bf16 bf16x8;
typedef __attribute__((ext_vector_type(2)))  bf16 bf16x2;
typedef __attribute__((ext_vector_type(8)))  float v8f;

union FragBF16 { v16bf v; bf16x8 h[2]; };

static __device__ __forceinline__ bf16x8 cvt8(float4 a, float4 b) {
    bf16x8 r;
    r[0] = (bf16)a.x; r[1] = (bf16)a.y; r[2] = (bf16)a.z; r[3] = (bf16)a.w;
    r[4] = (bf16)b.x; r[5] = (bf16)b.y; r[6] = (bf16)b.z; r[7] = (bf16)b.w;
    return r;
}

// ---------------------------------------------------------------------------
// Kernel 1: btp[k][n] = e1[n] @ W[k] @ e2[n]    (bf16 WMMA, f32 accumulate)
//
// Block = 256 threads = 8 wave32s; block owns (k, 256-row batch tile); each
// wave owns TWO 16-row M-tiles (B fragment reused -> 1 ds_load_b128 / WMMA),
// with B fragments software-pipelined 2 deep so DS latency overlaps WMMAs.
// A fragments are REGISTER-resident per 256-wide d-quarter (128 VGPRs).
// W / e2 panels are double-buffered in LDS; W staged via float4 loads +
// v_cvt_pk_bf16_f32 + b32 stores; e2 panel staged with the CDNA5 async
// global->LDS scatter path (ASYNCcnt) doing the transpose for free.
// Grid = (B/256) * K = 16*32 = 512.
// ---------------------------------------------------------------------------
__global__ __launch_bounds__(256, 1)
void ntl_btp_kernel(const float* __restrict__ e1,
                    const float* __restrict__ e2,
                    const float* __restrict__ W,
                    float* __restrict__ btp) {
    // Panels: 16 e-cols x 256 d (or 256 rows), padded stride vs bank conflicts
    __shared__ __align__(16) bf16  lB[2][16][264];    // W^T quarter-panel, bf16
    __shared__ __align__(16) float le2t[2][16][264];  // e2^T panel, f32
    __shared__ float lAcc[256];

    const int k  = blockIdx.x & (NK - 1);
    const int nb = (blockIdx.x >> 5) * 256;          // batch-row base
    const int t    = threadIdx.x;
    const int lane = t & 31;
    const int wv   = t >> 5;                          // wave id 0..7
    const int hi   = lane >> 4;                       // lane half-group
    const int ncol = lane & 15;                       // B/C fragment column

    // W staging mapping: 4-col group + 2 adjacent d-rows per unit
    const int j0 = (t & 3) * 4;                       // e-col group base (0,4,8,12)
    const int db = (t >> 2) * 2;                      // d-pair base (0..126), +128 for unit 1

    lAcc[t] = 0.0f;

    const float* Wk = W + (size_t)k * ND * ND;

    // --- staging helpers -------------------------------------------------
    // W panel (q, ec) -> registers: 2 units x 2 rows x float4 (coalesced 64B)
    auto load_w = [&](int q, int ec, float4* wr) {
#pragma unroll
        for (int u = 0; u < 2; ++u) {
            const float* src = Wk + (size_t)(q * 256 + db + 128 * u) * ND + ec * 16 + j0;
            wr[u * 2 + 0] = *(const float4*)src;
            wr[u * 2 + 1] = *(const float4*)(src + ND);
        }
    };
    auto store_w = [&](int buf, const float4* wr) {
#pragma unroll
        for (int u = 0; u < 2; ++u) {
            const float a[4] = {wr[u*2].x,   wr[u*2].y,   wr[u*2].z,   wr[u*2].w};
            const float b[4] = {wr[u*2+1].x, wr[u*2+1].y, wr[u*2+1].z, wr[u*2+1].w};
#pragma unroll
            for (int jj = 0; jj < 4; ++jj) {
                bf16x2 p;                              // d-pair -> one b32 store
                p[0] = (bf16)a[jj];
                p[1] = (bf16)b[jj];
                *(bf16x2*)&lB[buf][j0 + jj][db + 128 * u] = p;
            }
        }
    };
    // e2 panel transpose-scatter via async global->LDS (tracked by ASYNCcnt)
    auto stage_e2_async = [&](int buf, int ec) {
#pragma unroll
        for (int i = 0; i < 16; ++i) {
            const int gidx = i * 256 + t;
            const int m = gidx >> 4, j = gidx & 15;
            unsigned ldsoff = (unsigned)(uintptr_t)&le2t[buf][j][m];
            const float* ga = e2 + (size_t)(nb + m) * ND + ec * 16 + j;
            asm volatile("global_load_async_to_lds_b32 %0, %1, off"
                         :: "v"(ldsoff), "v"(ga) : "memory");
        }
    };

    // --- register-resident A fragments: 2 M-tiles x 8 K-steps ------------
    FragBF16 a0[8], a1[8];
    auto load_a_quarter = [&](int q) {
        const float* r0 = e1 + (size_t)(nb + wv * 32 + (lane & 15)) * ND + q * 256 + hi * 8;
        const float* r1 = r0 + (size_t)16 * ND;
#pragma unroll
        for (int dc = 0; dc < 8; ++dc) {
            const float4* p0 = (const float4*)(r0 + dc * 32);
            const float4* p1 = (const float4*)(r0 + dc * 32 + 16);
            a0[dc].h[0] = cvt8(p0[0], p0[1]);      // K {0..7}/{8..15} (by hi)
            a0[dc].h[1] = cvt8(p1[0], p1[1]);      // K {16..23}/{24..31}
            const float4* q0 = (const float4*)(r1 + dc * 32);
            const float4* q1 = (const float4*)(r1 + dc * 32 + 16);
            a1[dc].h[0] = cvt8(q0[0], q0[1]);
            a1[dc].h[1] = cvt8(q1[0], q1[1]);
        }
    };

    // B fragment loader (1 ds_load_b128 pair, shared by both M-tiles)
    auto load_b = [&](int buf, int dc, FragBF16& dst) {
        const int kb = dc * 32 + hi * 8;    // lanes<16: K 0..7,16..23; >=16: K 8..15,24..31
        dst.h[0] = *(const bf16x8*)&lB[buf][ncol][kb];
        dst.h[1] = *(const bf16x8*)&lB[buf][ncol][kb + 16];
    };

    float acc0[8], acc1[8];
#pragma unroll
    for (int i = 0; i < 8; ++i) { acc0[i] = 0.0f; acc1[i] = 0.0f; }

    // --- prologue: stage (q=0, ec=0) into buffer 0 ------------------------
    {
        float4 wr[4];
        load_w(0, 0, wr);
        stage_e2_async(0, 0);
        store_w(0, wr);
        asm volatile("s_wait_asynccnt 0x0" ::: "memory");
    }
    __syncthreads();

    // --- main loop: 4 d-quarters x 64 e-chunks = 256 iterations -----------
    for (int idx = 0; idx < 256; ++idx) {
        const int buf = idx & 1;
        if ((idx & 63) == 0) load_a_quarter(idx >> 6);

        // issue next panel's global loads early (hidden behind WMMA burst)
        float4 wr[4];
        const bool have_next = (idx + 1 < 256);
        if (have_next) {
            const int nidx = idx + 1;
            load_w(nidx >> 6, nidx & 63, wr);
            stage_e2_async(buf ^ 1, nidx & 63);
        }
        if (idx + 2 < 256) {
            const int pidx = idx + 2;
            __builtin_prefetch(Wk + (size_t)(((pidx >> 6) * 256) + t) * ND
                                  + (pidx & 63) * 16, 0, 3);
        }

        // ---- 16 WMMAs; B fragments pipelined 2 deep (3-slot rotation) ----
        v8f c0 = {0.f,0.f,0.f,0.f,0.f,0.f,0.f,0.f};
        v8f c1 = {0.f,0.f,0.f,0.f,0.f,0.f,0.f,0.f};
        {
            FragBF16 bf3[3];
            load_b(buf, 0, bf3[0]);
            load_b(buf, 1, bf3[1]);
#pragma unroll
            for (int dc = 0; dc < 8; ++dc) {
                if (dc + 2 < 8) load_b(buf, dc + 2, bf3[(dc + 2) % 3]);
                const v16bf bv = bf3[dc % 3].v;
                c0 = __builtin_amdgcn_wmma_f32_16x16x32_bf16(
                         false, a0[dc].v, false, bv, (short)0, c0, false, false);
                c1 = __builtin_amdgcn_wmma_f32_16x16x32_bf16(
                         false, a1[dc].v, false, bv, (short)0, c1, false, false);
            }
        }

        // ---- weight partial tiles by e2 (C-layout: col=ncol, rows base+r)
        {
            const float* p0 = &le2t[buf][ncol][wv * 32 + 8 * hi];
            const float* p1 = &le2t[buf][ncol][wv * 32 + 16 + 8 * hi];
            float4 fa0 = *(const float4*)p0, fb0 = *(const float4*)(p0 + 4);
            float4 fa1 = *(const float4*)p1, fb1 = *(const float4*)(p1 + 4);
            acc0[0] += c0[0]*fa0.x; acc0[1] += c0[1]*fa0.y;
            acc0[2] += c0[2]*fa0.z; acc0[3] += c0[3]*fa0.w;
            acc0[4] += c0[4]*fb0.x; acc0[5] += c0[5]*fb0.y;
            acc0[6] += c0[6]*fb0.z; acc0[7] += c0[7]*fb0.w;
            acc1[0] += c1[0]*fa1.x; acc1[1] += c1[1]*fa1.y;
            acc1[2] += c1[2]*fa1.z; acc1[3] += c1[3]*fa1.w;
            acc1[4] += c1[4]*fb1.x; acc1[5] += c1[5]*fb1.y;
            acc1[6] += c1[6]*fb1.z; acc1[7] += c1[7]*fb1.w;
        }

        // ---- commit staged W to the other buffer, fence async e2 ---------
        if (have_next) store_w(buf ^ 1, wr);
        asm volatile("s_wait_asynccnt 0x0" ::: "memory");
        __syncthreads();
    }

    // ---- reduce across the 16 N-columns -> per-row btp -------------------
#pragma unroll
    for (int r = 0; r < 8; ++r) {
        atomicAdd(&lAcc[wv * 32 +      hi * 8 + r], acc0[r]);
        atomicAdd(&lAcc[wv * 32 + 16 + hi * 8 + r], acc1[r]);
    }
    __syncthreads();
    btp[(size_t)k * NB + nb + t] = lAcc[t];
}

// ---------------------------------------------------------------------------
// Kernel 2: out[n][j] = tanh( btp[n>>7][(n&127)*32 + j] + ff[n][j] + sum(b) )
// ff[n][j] = e1[n]·V[0:D, j] + e2[n]·V[D:2D, j].  One block per batch row.
// ---------------------------------------------------------------------------
__global__ __launch_bounds__(256, 4)
void ntl_combine_kernel(const float* __restrict__ e1,
                        const float* __restrict__ e2,
                        const float* __restrict__ V,
                        const float* __restrict__ b,
                        const float* __restrict__ btp,
                        float* __restrict__ out) {
    __shared__ float red[8][32];
    __shared__ float bsum;

    const int n = blockIdx.x;
    const int t = threadIdx.x;
    const int j   = t & 31;
    const int seg = t >> 5;

    if (t == 0) bsum = 0.0f;
    __syncthreads();

    const float* e1r = e1 + (size_t)n * ND;
    const float* e2r = e2 + (size_t)n * ND;

    float p = 0.0f;
    const int d0 = seg * (ND / 8);
#pragma unroll 4
    for (int d = d0; d < d0 + ND / 8; ++d) {
        p += e1r[d] * V[(size_t)d * NK + j];
        p += e2r[d] * V[(size_t)(ND + d) * NK + j];
    }
    red[seg][j] = p;

    float pb = b[t] + b[t + 256] + b[t + 512] + b[t + 768];
    atomicAdd(&bsum, pb);
    __syncthreads();

    if (t < 32) {
        float ff = 0.0f;
#pragma unroll
        for (int s = 0; s < 8; ++s) ff += red[s][t];
        const int kidx = n >> 7;                  // reshape((K,B)->(B,K)) interleave
        const int bidx = (n & 127) * NK + t;
        const float v = btp[(size_t)kidx * NB + bidx] + ff + bsum;
        out[(size_t)n * NK + t] = tanhf(v);
    }
}

// ---------------------------------------------------------------------------
extern "C" void kernel_launch(void* const* d_in, const int* in_sizes, int n_in,
                              void* d_out, int out_size, void* d_ws, size_t ws_size,
                              hipStream_t stream) {
    const float* e1 = (const float*)d_in[0];   // (B, D)
    const float* e2 = (const float*)d_in[1];   // (B, D)
    const float* W  = (const float*)d_in[2];   // (K, D, D)
    const float* V  = (const float*)d_in[3];   // (2D, K)
    const float* b  = (const float*)d_in[4];   // (D,)
    float* out = (float*)d_out;                // (B, K)
    float* btp = (float*)d_ws;                 // K*B floats = 512 KB scratch

    ntl_btp_kernel<<<dim3((NB / 256) * NK), 256, 0, stream>>>(e1, e2, W, btp);
    ntl_combine_kernel<<<dim3(NB), 256, 0, stream>>>(e1, e2, V, b, btp, out);
}
